// Q2A_85272280695393
// MI455X (gfx1250) — matrix operand order repaired and checked by reading, not compiled, not run on hardware
//
#include <hip/hip_runtime.h>
#include <hip/hip_bf16.h>

// ---------------------------------------------------------------------------
// Problem constants (from reference): D=768 H=12 S=768 M=1 B=8 TV=512 TS=128
// STEPS=8 A=16, DP=(3+M)*D=3072, dh=64, scale=1/8.
// ---------------------------------------------------------------------------

typedef unsigned short u16;

__device__ __forceinline__ u16 f2bf(float f) {
  unsigned int u = __float_as_uint(f);
  unsigned int r = u + 0x7FFFu + ((u >> 16) & 1u);   // round-to-nearest-even
  return (u16)(r >> 16);
}
__device__ __forceinline__ float bf2f(u16 h) {
  return __uint_as_float(((unsigned int)h) << 16);
}

typedef __attribute__((ext_vector_type(16))) __bf16 v16bf;
typedef __attribute__((ext_vector_type(8)))  float  v8f;

union FragU { uint4 q[2]; v16bf v; };

// -------- gfx1250 async global->LDS copies (ASYNCcnt) via inline asm -------
__device__ __forceinline__ void async_ld_b128(unsigned lds, const void* g) {
  asm volatile("global_load_async_to_lds_b128 %0, %1, off"
               :: "v"(lds), "v"((unsigned long long)(uintptr_t)g) : "memory");
}
__device__ __forceinline__ void async_ld_b64(unsigned lds, const void* g) {
  asm volatile("global_load_async_to_lds_b64 %0, %1, off"
               :: "v"(lds), "v"((unsigned long long)(uintptr_t)g) : "memory");
}
__device__ __forceinline__ void wait_async5() {
  asm volatile("s_wait_asynccnt 0x5" ::: "memory");
}
__device__ __forceinline__ void wait_async0() {
  asm volatile("s_wait_asynccnt 0x0" ::: "memory");
}

// ------------------------------ WMMA GEMM (direct) -------------------------
// C[M,N] = act(A[M,K] @ Bpacked + bias), A row-major bf16 (lda elements),
// Bp pre-packed per ISA B-fragment layout: [K/32][N/16][lane 0..31][16 bf16],
// where lane l elem i holds B[k=(i/8)*16+(l/16)*8+i%8][n=l%16] of the 32x16 tile.
// A fragment per ISA 16-bit A 16x32 layout: lane l (m=l&15, half=l>>4) holds
// k = half*8+0..7 (elems 0-7) and k = 16+half*8+0..7 (elems 8-15).
// One wave per 16x16 C tile; used for odd shapes (M=16, N=64/128).
// ---------------------------------------------------------------------------

__global__ __launch_bounds__(256) void gemm_bf16_wmma(
    const u16* __restrict__ A, int lda, long sAy, long sAz,
    const u16* __restrict__ Bp, long sBy, long sBz,
    const float* __restrict__ bias,
    float* __restrict__ Cf, int ldcf, long sCfy, long sCfz,
    u16* __restrict__ Cb, int ldcb, long sCby, long sCbz,
    int M, int N, int K, int relu)
{
  A  += (long)blockIdx.y * sAy + (long)blockIdx.z * sAz;
  Bp += (long)blockIdx.y * sBy + (long)blockIdx.z * sBz;
  if (Cf) Cf += (long)blockIdx.y * sCfy + (long)blockIdx.z * sCfz;
  if (Cb) Cb += (long)blockIdx.y * sCby + (long)blockIdx.z * sCbz;

  const int ntN   = N >> 4;
  const int tiles = (M >> 4) * ntN;
  const int tid   = blockIdx.x * 8 + (threadIdx.x >> 5);
  if (tid >= tiles) return;
  const int mt = tid / ntN, nt = tid % ntN;
  const int lane = threadIdx.x & 31;
  const int half = lane >> 4;

  const u16* aptr = A + (size_t)((mt << 4) + (lane & 15)) * lda + half * 8;
  const u16* bptr = Bp + (size_t)nt * 512 + lane * 16;
  const size_t bstep = (size_t)ntN * 512;

  v8f acc = {};
  for (int kk = 0; kk < K; kk += 32) {
    FragU fa, fb;
    fa.q[0] = *(const uint4*)(aptr);
    fa.q[1] = *(const uint4*)(aptr + 16);
    fb.q[0] = *(const uint4*)(bptr);
    fb.q[1] = *(const uint4*)(bptr + 8);
    __builtin_prefetch(aptr + 32, 0, 1);        // global_prefetch_b8
    __builtin_prefetch(bptr + bstep, 0, 1);
    acc = __builtin_amdgcn_wmma_f32_16x16x32_bf16(
        false, fa.v, false, fb.v, (short)0, acc, false, false);
    aptr += 32;
    bptr += bstep;
  }

  const int col   = (nt << 4) + (lane & 15);
  const int rbase = (mt << 4) + (half << 3);
  const float bv  = bias ? bias[col] : 0.0f;
#pragma unroll
  for (int r = 0; r < 8; ++r) {
    float v = acc[r] + bv;
    if (relu) v = fmaxf(v, 0.0f);
    const int row = rbase + r;
    if (Cf) Cf[(size_t)row * ldcf + col] = v;
    if (Cb) Cb[(size_t)row * ldcb + col] = f2bf(v);
  }
}

// --------------------- WMMA GEMM with async-LDS staging --------------------
// Requires M%32==0, N%256==0, K%32==0. Block = 256 threads = 8 waves laid out
// 2(M) x 4(N); block tile = 32 x 256. Per K-chunk (32), A panel (32x32) and
// B panel (16 packed tiles = 16KB) are staged into LDS with
// GLOBAL_LOAD_ASYNC_TO_LDS_{B64,B128} (double buffered, ASYNCcnt in-order
// completion => s_wait_asynccnt 5 retires the previous stage only), then each
// wave runs 4 back-to-back WMMAs sharing one A fragment read via ds_load_b128.
// A panel row pitch = 40 elems (80B) so the 16 per-half row reads hit 16
// distinct bank groups. LDS: 2*(2.5KB + 16KB) = 37KB of the 320KB/WGP.
// ---------------------------------------------------------------------------

__global__ __launch_bounds__(256) void gemm_bf16_wmma_lds(
    const u16* __restrict__ A, int lda, long sAy, long sAz,
    const u16* __restrict__ Bp, long sBy, long sBz,
    const float* __restrict__ bias,
    float* __restrict__ Cf, int ldcf, long sCfy, long sCfz,
    u16* __restrict__ Cb, int ldcb, long sCby, long sCbz,
    int M, int N, int K, int relu)
{
  __shared__ __align__(16) u16 As[2][32 * 40];
  __shared__ __align__(16) u16 Bs[2][16 * 512];

  A  += (long)blockIdx.y * sAy + (long)blockIdx.z * sAz;
  Bp += (long)blockIdx.y * sBy + (long)blockIdx.z * sBz;
  if (Cf) Cf += (long)blockIdx.y * sCfy + (long)blockIdx.z * sCfz;
  if (Cb) Cb += (long)blockIdx.y * sCby + (long)blockIdx.z * sCbz;

  const int nb256    = N >> 8;
  const int ntBase   = (blockIdx.x % nb256) << 4;   // first 16-wide tile
  const int mrowBase = (blockIdx.x / nb256) << 5;   // first row
  const int ntN      = N >> 4;
  const size_t bstep = (size_t)ntN * 512;

  const int t = threadIdx.x;
  const int w = t >> 5, lane = t & 31, half = lane >> 4;

  // staging addresses (uniform 5 async ops per wave per stage)
  const unsigned asBase = (unsigned)(uintptr_t)&As[0][0];
  const unsigned bsBase = (unsigned)(uintptr_t)&Bs[0][0];
  const int ar = t >> 3, ac = (t & 7) << 2;               // A: 32 rows x 32 cols
  const unsigned aLds = asBase + (unsigned)(ar * 80 + ac * 2);
  const unsigned bLds = bsBase + (unsigned)(t * 16);
  const u16* aSrcRow  = A + (size_t)(mrowBase + ar) * lda + ac;
  const u16* bSrcBase = Bp + (size_t)ntBase * 512 + (size_t)t * 8;

  auto stage = [&](int buf, int kk) {
    const u16* bsrc = bSrcBase + (size_t)(kk >> 5) * bstep;
    const unsigned bufA = (unsigned)buf * (32u * 40u * 2u);
    const unsigned bufB = (unsigned)buf * (16u * 512u * 2u);
#pragma unroll
    for (int i = 0; i < 4; ++i)
      async_ld_b128(bLds + bufB + (unsigned)i * 4096u, bsrc + i * 2048);
    async_ld_b64(aLds + bufA, aSrcRow + kk);
  };

  stage(0, 0);

  v8f acc0 = {}, acc1 = {}, acc2 = {}, acc3 = {};
  const int aoff = ((w >> 2) * 16 + (lane & 15)) * 40 + half * 8;
  const int boff = ((w & 3) << 2) * 512 + lane * 16;

  const int nchunk = K >> 5;
  for (int c = 0; c < nchunk; ++c) {
    const int buf = c & 1;
    if (c + 1 < nchunk) { stage(buf ^ 1, (c + 1) << 5); wait_async5(); }
    else                { wait_async0(); }
    __syncthreads();

    FragU fa;
    const u16* ap = &As[buf][aoff];
    fa.q[0] = *(const uint4*)(ap);
    fa.q[1] = *(const uint4*)(ap + 16);
    const u16* bb = &Bs[buf][boff];
    FragU fb0, fb1, fb2, fb3;
    fb0.q[0] = *(const uint4*)(bb);          fb0.q[1] = *(const uint4*)(bb + 8);
    fb1.q[0] = *(const uint4*)(bb + 512);    fb1.q[1] = *(const uint4*)(bb + 520);
    fb2.q[0] = *(const uint4*)(bb + 1024);   fb2.q[1] = *(const uint4*)(bb + 1032);
    fb3.q[0] = *(const uint4*)(bb + 1536);   fb3.q[1] = *(const uint4*)(bb + 1544);
    acc0 = __builtin_amdgcn_wmma_f32_16x16x32_bf16(false, fa.v, false, fb0.v, (short)0, acc0, false, false);
    acc1 = __builtin_amdgcn_wmma_f32_16x16x32_bf16(false, fa.v, false, fb1.v, (short)0, acc1, false, false);
    acc2 = __builtin_amdgcn_wmma_f32_16x16x32_bf16(false, fa.v, false, fb2.v, (short)0, acc2, false, false);
    acc3 = __builtin_amdgcn_wmma_f32_16x16x32_bf16(false, fa.v, false, fb3.v, (short)0, acc3, false, false);
    __syncthreads();
  }

  v8f accs[4] = {acc0, acc1, acc2, acc3};
  const int col0  = (ntBase + ((w & 3) << 2)) * 16 + (lane & 15);
  const int rbase = mrowBase + ((w >> 2) << 4) + (half << 3);
#pragma unroll
  for (int j = 0; j < 4; ++j) {
    const int col  = col0 + j * 16;
    const float bv = bias ? bias[col] : 0.0f;
#pragma unroll
    for (int r = 0; r < 8; ++r) {
      float v = accs[j][r] + bv;
      if (relu) v = fmaxf(v, 0.0f);
      const int row = rbase + r;
      if (Cf) Cf[(size_t)row * ldcf + col] = v;
      if (Cb) Cb[(size_t)row * ldcb + col] = f2bf(v);
    }
  }
}

// --------------------- B-operand packers (to fragment layout) --------------

__global__ void pack_b_f32(const float* __restrict__ src, long sK, long sN,
                           long sy, long sz,
                           u16* __restrict__ dst, long dy, long dz,
                           int K, int N)
{
  src += (long)blockIdx.y * sy + (long)blockIdx.z * sz;
  dst += (long)blockIdx.y * dy + (long)blockIdx.z * dz;
  const int  ntN   = N >> 4;
  const long total = (long)(K >> 5) * ntN * 512;
  for (long idx = (long)blockIdx.x * blockDim.x + threadIdx.x; idx < total;
       idx += (long)gridDim.x * blockDim.x) {
    long kt  = idx / ((long)ntN * 512);
    long rem = idx % ((long)ntN * 512);
    int  nt = (int)(rem / 512), e = (int)(rem % 512);
    int  l = e >> 4, i = e & 15;
    int  k = (int)(kt * 32) + ((i >> 3) << 4) + ((l >> 4) << 3) + (i & 7);
    int  n = nt * 16 + (l & 15);
    dst[idx] = f2bf(src[(long)k * sK + (long)n * sN]);
  }
}

__global__ void pack_b_bf16(const u16* __restrict__ src, long sK, long sN,
                            long sy, long sz,
                            u16* __restrict__ dst, long dy, long dz,
                            int K, int N)
{
  src += (long)blockIdx.y * sy + (long)blockIdx.z * sz;
  dst += (long)blockIdx.y * dy + (long)blockIdx.z * dz;
  const int  ntN   = N >> 4;
  const long total = (long)(K >> 5) * ntN * 512;
  for (long idx = (long)blockIdx.x * blockDim.x + threadIdx.x; idx < total;
       idx += (long)gridDim.x * blockDim.x) {
    long kt  = idx / ((long)ntN * 512);
    long rem = idx % ((long)ntN * 512);
    int  nt = (int)(rem / 512), e = (int)(rem % 512);
    int  l = e >> 4, i = e & 15;
    int  k = (int)(kt * 32) + ((i >> 3) << 4) + ((l >> 4) << 3) + (i & 7);
    int  n = nt * 16 + (l & 15);
    dst[idx] = src[(long)k * sK + (long)n * sN];
  }
}

// ----------------------------- small kernels -------------------------------

__global__ void cvt_f32_bf16(const float* __restrict__ s, u16* __restrict__ d, long n) {
  for (long i = (long)blockIdx.x * blockDim.x + threadIdx.x; i < n;
       i += (long)gridDim.x * blockDim.x)
    d[i] = f2bf(s[i]);
}

__global__ void build_qpad(const float* __restrict__ q, u16* __restrict__ qp) {
  int idx = blockIdx.x * 256 + threadIdx.x;
  if (idx >= 16 * 768) return;
  int r = idx / 768;
  qp[idx] = (r < 8) ? f2bf(q[idx]) : (u16)0;
}

// softmax over rows of length L on bf16 data (scale applied pre-exp)
__global__ void softmax_rows(const u16* __restrict__ src, u16* __restrict__ dst,
                             int L, float scale)
{
  __shared__ float red[256];
  const long row = blockIdx.x;
  const u16* s = src + row * (long)L;
  u16*       d = dst + row * (long)L;
  float mx = -3.0e38f;
  for (int i = threadIdx.x; i < L; i += blockDim.x)
    mx = fmaxf(mx, bf2f(s[i]) * scale);
  red[threadIdx.x] = mx; __syncthreads();
  for (int o = 128; o; o >>= 1) {
    if ((int)threadIdx.x < o) red[threadIdx.x] = fmaxf(red[threadIdx.x], red[threadIdx.x + o]);
    __syncthreads();
  }
  mx = red[0]; __syncthreads();
  float sum = 0.f;
  for (int i = threadIdx.x; i < L; i += blockDim.x)
    sum += __expf(bf2f(s[i]) * scale - mx);
  red[threadIdx.x] = sum; __syncthreads();
  for (int o = 128; o; o >>= 1) {
    if ((int)threadIdx.x < o) red[threadIdx.x] += red[threadIdx.x + o];
    __syncthreads();
  }
  const float inv = 1.f / red[0];
  for (int i = threadIdx.x; i < L; i += blockDim.x)
    d[i] = f2bf(__expf(bf2f(s[i]) * scale - mx) * inv);
}

// w[b,q,t] = mean over 12 heads of probs[b,h,q,t]  (q=t=128)
__global__ void mean_heads(const u16* __restrict__ pr, u16* __restrict__ wout) {
  long idx = (long)blockIdx.x * 256 + threadIdx.x;
  if (idx >= 8L * 128 * 128) return;
  long b = idx / (128 * 128), qt = idx % (128 * 128);
  float s = 0.f;
#pragma unroll
  for (int h = 0; h < 12; ++h) s += bf2f(pr[(b * 12 + h) * 16384 + qt]);
  wout[idx] = f2bf(s * (1.0f / 12.0f));
}

// qa = qm[b] + at_m[row]; write fp32 into inp slot 3 (cols 1536..2303) + bf16
__global__ void add_qa(const float* __restrict__ qm, const float* __restrict__ atm,
                       float* __restrict__ inp, u16* __restrict__ qa)
{
  long idx = (long)blockIdx.x * 256 + threadIdx.x;
  if (idx >= 1024L * 768) return;
  long row = idx / 768; int d = (int)(idx % 768); int b = (int)(row >> 7);
  float v = qm[(long)b * 768 + d] + atm[idx];
  inp[row * 3072 + 1536 + d] = v;
  qa[idx] = f2bf(v);
}

__global__ void init_state(const float* __restrict__ s0, float* __restrict__ sf,
                           u16* __restrict__ sb)
{
  int idx = blockIdx.x * 256 + threadIdx.x;
  if (idx >= 16 * 768) return;
  float x = s0[idx % 768];
  sf[idx] = x;
  sb[idx] = f2bf(x);
}

// GRU gates: r=sig(ir+hr) z=sig(iz+hz) n=tanh(in+r*hn); h=(1-z)*n+z*state
__global__ void gru_gates(const float* __restrict__ gi, const float* __restrict__ gh,
                          const float* __restrict__ state_f, u16* __restrict__ h_bf,
                          int step)
{
  long idx = (long)blockIdx.x * 256 + threadIdx.x;
  if (idx >= 8L * 16 * 768) return;
  int  d  = (int)(idx % 768);
  long ba = idx / 768;
  int  a  = (int)(ba % 16), b = (int)(ba / 16);
  long row = (long)b * 128 + (long)step * 16 + a;
  const float* g  = gi + row * 2304;
  const float* gb = gh + (long)b * 2304;
  float r = 1.f / (1.f + __expf(-(g[d]        + gb[d])));
  float z = 1.f / (1.f + __expf(-(g[768 + d]  + gb[768 + d])));
  float n = tanhf(g[1536 + d] + r * gb[1536 + d]);
  float hv = (1.f - z) * n + z * state_f[(long)b * 768 + d];
  h_bf[((long)b * 16 + a) * 768 + d] = f2bf(hv);
}

// logits[row] = dot(p1[row,:], proj_w2) + b2  -> d_out[b*128+step*16+a]
__global__ void proj_logits(const u16* __restrict__ p1, const float* __restrict__ w2,
                            const float* __restrict__ b2, float* __restrict__ out,
                            int step)
{
  int row = blockIdx.x, lane = threadIdx.x;
  float s = 0.f;
  for (int i = lane; i < 768; i += 32) s += bf2f(p1[(long)row * 768 + i]) * w2[i];
  for (int o = 16; o; o >>= 1) s += __shfl_down(s, o, 32);
  if (lane == 0) {
    int b = row >> 4, a = row & 15;
    out[b * 128 + step * 16 + a] = s + b2[0];
  }
}

// argmax over A=16 logits per batch; new state = inp2 row (fp32 + bf16 copies)
__global__ void argmax_update(const float* __restrict__ out, const float* __restrict__ inp2f,
                              float* __restrict__ state_f, u16* __restrict__ state_bf,
                              int step)
{
  int b = blockIdx.x, lane = threadIdx.x;
  float v  = (lane < 16) ? out[b * 128 + step * 16 + lane] : -3.4e38f;
  int   id = (lane < 16) ? lane : 0;
  for (int o = 16; o; o >>= 1) {
    float ov = __shfl_down(v, o, 32);
    int   oi = __shfl_down(id, o, 32);
    if (ov > v) { v = ov; id = oi; }
  }
  id = __shfl(id, 0, 32);
  long srow = (long)b * 128 + (long)step * 16 + id;
  for (int i = lane; i < 768; i += 32) {
    float x = inp2f[srow * 768 + i];
    state_f[(long)b * 768 + i]  = x;
    state_bf[(long)b * 768 + i] = f2bf(x);
  }
}

// ------------------------------- host side ---------------------------------

extern "C" void kernel_launch(void* const* d_in, const int* in_sizes, int n_in,
                              void* d_out, int out_size, void* d_ws, size_t ws_size,
                              hipStream_t stream)
{
  (void)in_sizes; (void)n_in; (void)out_size; (void)ws_size;
  const float* video    = (const float*)d_in[0];
  const float* script   = (const float*)d_in[1];
  const float* question = (const float*)d_in[2];
  const float* a_texts  = (const float*)d_in[3];
  const float* a_btns   = (const float*)d_in[4];
  const float *v_w1 = (const float*)d_in[5],  *v_b1 = (const float*)d_in[6];
  const float *v_w2 = (const float*)d_in[7],  *v_b2 = (const float*)d_in[8];
  const float *t_w1 = (const float*)d_in[9],  *t_b1 = (const float*)d_in[10];
  const float *t_w2 = (const float*)d_in[11], *t_b2 = (const float*)d_in[12];
  const float *pre_w1 = (const float*)d_in[13], *pre_b1 = (const float*)d_in[14];
  const float *pre_w2 = (const float*)d_in[15], *pre_b2 = (const float*)d_in[16];
  const float *s2v_win = (const float*)d_in[17], *s2v_bin = (const float*)d_in[18];
  const float *s2v_wout = (const float*)d_in[19], *s2v_bout = (const float*)d_in[20];
  const float *qa_win = (const float*)d_in[21], *qa_bin = (const float*)d_in[22];
  const float *qa_wout = (const float*)d_in[23], *qa_bout = (const float*)d_in[24];
  const float *gru_wih = (const float*)d_in[25], *gru_whh = (const float*)d_in[26];
  const float *gru_bih = (const float*)d_in[27], *gru_bhh = (const float*)d_in[28];
  const float *proj_w1 = (const float*)d_in[29], *proj_b1 = (const float*)d_in[30];
  const float *proj_w2 = (const float*)d_in[31], *proj_b2 = (const float*)d_in[32];
  const float *state0  = (const float*)d_in[33];
  float* out = (float*)d_out;

  // ---- workspace bump allocator ----
  char* wsb = (char*)d_ws; size_t off = 0;
  auto alloc = [&](size_t bytes) -> void* {
    void* p = wsb + off;
    off += (bytes + 255) & ~(size_t)255;
    return p;
  };
  auto B16 = [&](size_t n) { return (u16*)alloc(n * 2); };
  auto F32 = [&](size_t n) { return (float*)alloc(n * 4); };

  // weights (packed bf16 B-fragment layout)
  u16 *pvw1 = B16(768*768), *pvw2 = B16(768*768), *ptw1 = B16(768*768), *ptw2 = B16(768*768);
  u16 *ps2q = B16(768*768), *ps2k = B16(768*768), *ps2v = B16(768*768), *ps2o = B16(768*768);
  u16 *pqaq = B16(768*768), *pqak = B16(768*768), *pqav = B16(768*768), *pqao = B16(768*768);
  u16 *ppre1 = B16((size_t)3072*3072), *ppre2 = B16((size_t)3072*768);
  u16 *pgih = B16((size_t)768*2304), *pghh = B16((size_t)768*2304), *pprj1 = B16(768*768);

  // activations
  u16 *vid_bf = B16((size_t)4096*768), *scr_bf = B16((size_t)1024*768);
  u16 *at_bf  = B16((size_t)1024*768), *ab_bf  = B16((size_t)1024*768), *qpad = B16(16*768);
  u16 *vh1_bf = B16((size_t)4096*768), *v_bf   = B16((size_t)4096*768);
  u16 *sch_bf = B16((size_t)1024*768), *sc_bf  = B16((size_t)1024*768);
  u16 *qh1_bf = B16((size_t)1024*768);
  u16 *kh_bf  = B16((size_t)4096*768), *vhh_bf = B16((size_t)4096*768);
  u16 *kp1 = B16((size_t)8*12*32768), *vp1 = B16((size_t)8*12*32768);
  u16 *sc1_bf = B16((size_t)8*12*65536), *pr1_bf = B16((size_t)8*12*65536);
  u16 *vattin_bf = B16((size_t)1024*768), *vatt_bf = B16((size_t)1024*768);
  u16 *qmh_bf = B16(16*768);
  u16 *ath_bf = B16((size_t)1024*768), *abh_bf = B16((size_t)1024*768);
  u16 *qa_bf  = B16((size_t)1024*768);
  u16 *qh2_bf = B16((size_t)1024*768), *k2_bf = B16((size_t)1024*768), *v2_bf = B16((size_t)1024*768);
  u16 *kp2 = B16((size_t)8*12*8192), *vp2 = B16((size_t)8*12*8192);
  u16 *sc2_bf = B16((size_t)8*12*16384), *pr2_bf = B16((size_t)8*12*16384);
  u16 *w_bf = B16((size_t)8*16384), *qsin_bf = B16((size_t)1024*768);
  u16 *vattp = B16((size_t)8*98304);
  u16 *inp_bf = B16((size_t)1024*3072), *preh_bf = B16((size_t)1024*3072);
  u16 *inp2_bf = B16((size_t)1024*768);
  u16 *state_bf = B16(16*768), *h_bf = B16((size_t)128*768), *p1_bf = B16((size_t)128*768);

  float *qm_f  = F32(16*768),  *atm_f = F32((size_t)1024*768);
  float *inp_f = F32((size_t)1024*3072), *inp2_f = F32((size_t)1024*768);
  float *gi_f  = F32((size_t)1024*2304), *gh_f = F32(16*2304), *state_f = F32(16*768);

  // ---- launch helpers ----
  auto G = [&](const u16* A, int lda, long sAy, long sAz,
               const u16* Bp, long sBy, long sBz, const float* bias,
               float* Cf, int ldcf, long sCfy, long sCfz,
               u16* Cb, int ldcb, long sCby, long sCbz,
               int M, int N, int K, int relu, int gy, int gz) {
    int tiles = (M / 16) * (N / 16);
    dim3 grid((tiles + 7) / 8, gy, gz);
    gemm_bf16_wmma<<<grid, 256, 0, stream>>>(A, lda, sAy, sAz, Bp, sBy, sBz, bias,
                                             Cf, ldcf, sCfy, sCfz, Cb, ldcb, sCby, sCbz,
                                             M, N, K, relu);
  };
  // async-LDS path: M%32==0, N%256==0
  auto GA = [&](const u16* A, int lda, long sAy, long sAz,
                const u16* Bp, long sBy, long sBz, const float* bias,
                float* Cf, int ldcf, long sCfy, long sCfz,
                u16* Cb, int ldcb, long sCby, long sCbz,
                int M, int N, int K, int relu, int gy, int gz) {
    dim3 grid((M / 32) * (N / 256), gy, gz);
    gemm_bf16_wmma_lds<<<grid, 256, 0, stream>>>(A, lda, sAy, sAz, Bp, sBy, sBz, bias,
                                                 Cf, ldcf, sCfy, sCfz, Cb, ldcb, sCby, sCbz,
                                                 M, N, K, relu);
  };
  auto PF = [&](const float* src, long sK, long sN, int K, int N, u16* dst) {
    long total = (long)K * N;
    int blocks = (int)((total + 255) / 256); if (blocks > 4096) blocks = 4096;
    pack_b_f32<<<dim3(blocks, 1, 1), 256, 0, stream>>>(src, sK, sN, 0, 0, dst, 0, 0, K, N);
  };
  auto PB = [&](const u16* src, long sK, long sN, long sy, long sz,
                u16* dst, long dy, long dz, int K, int N, int gy, int gz) {
    long total = (long)K * N;
    int blocks = (int)((total + 255) / 256); if (blocks > 2048) blocks = 2048;
    pack_b_bf16<<<dim3(blocks, gy, gz), 256, 0, stream>>>(src, sK, sN, sy, sz, dst, dy, dz, K, N);
  };
  auto CVT = [&](const float* s, u16* d, long n) {
    cvt_f32_bf16<<<dim3(2048), 256, 0, stream>>>(s, d, n);
  };

  // ---- weight packing (B[k,n] = W[n,k] -> sK=1, sN=K) ----
  PF(v_w1, 1, 768, 768, 768, pvw1);   PF(v_w2, 1, 768, 768, 768, pvw2);
  PF(t_w1, 1, 768, 768, 768, ptw1);   PF(t_w2, 1, 768, 768, 768, ptw2);
  PF(s2v_win + 0*768*768, 1, 768, 768, 768, ps2q);
  PF(s2v_win + 1*768*768, 1, 768, 768, 768, ps2k);
  PF(s2v_win + 2*768*768, 1, 768, 768, 768, ps2v);
  PF(s2v_wout, 1, 768, 768, 768, ps2o);
  PF(qa_win + 0*768*768, 1, 768, 768, 768, pqaq);
  PF(qa_win + 1*768*768, 1, 768, 768, 768, pqak);
  PF(qa_win + 2*768*768, 1, 768, 768, 768, pqav);
  PF(qa_wout, 1, 768, 768, 768, pqao);
  PF(pre_w1, 1, 3072, 3072, 3072, ppre1);
  PF(pre_w2, 1, 3072, 3072, 768, ppre2);
  PF(gru_wih, 1, 768, 768, 2304, pgih);
  PF(gru_whh, 1, 768, 768, 2304, pghh);
  PF(proj_w1, 1, 768, 768, 768, pprj1);

  // ---- input conversions ----
  CVT(video,  vid_bf, 4096L*768);
  CVT(script, scr_bf, 1024L*768);
  CVT(a_texts, at_bf, 1024L*768);
  CVT(a_btns,  ab_bf, 1024L*768);
  build_qpad<<<dim3(48), 256, 0, stream>>>(question, qpad);

  // ---- video / script MLPs ----
  GA(vid_bf,768,0,0, pvw1,0,0, v_b1, 0,0,0,0, vh1_bf,768,0,0, 4096,768,768, 1, 1,1);
  GA(vh1_bf,768,0,0, pvw2,0,0, v_b2, 0,0,0,0, v_bf,768,0,0,   4096,768,768, 0, 1,1);
  GA(scr_bf,768,0,0, ptw1,0,0, t_b1, 0,0,0,0, sch_bf,768,0,0, 1024,768,768, 1, 1,1);
  GA(sch_bf,768,0,0, ptw2,0,0, t_b2, 0,0,0,0, sc_bf,768,0,0,  1024,768,768, 0, 1,1);

  // ---- script->video MHA (q=sc, k=v, v=v), 12 heads of 64 ----
  GA(sc_bf,768,0,0, ps2q,0,0, s2v_bin+0,    0,0,0,0, qh1_bf,768,0,0, 1024,768,768, 0, 1,1);
  GA(v_bf, 768,0,0, ps2k,0,0, s2v_bin+768,  0,0,0,0, kh_bf,768,0,0,  4096,768,768, 0, 1,1);
  GA(v_bf, 768,0,0, ps2v,0,0, s2v_bin+1536, 0,0,0,0, vhh_bf,768,0,0, 4096,768,768, 0, 1,1);
  // pack K^T (B[k=d][n=t]) and V (B[k=t][n=d]) per (head=y, batch=z)
  PB(kh_bf, 1, 768, 64, 512L*768, kp1, 32768, 12L*32768, 64, 512, 12, 8);
  PB(vhh_bf, 768, 1, 64, 512L*768, vp1, 32768, 12L*32768, 512, 64, 12, 8);
  // scores [128,512] per (b,h), then softmax(1/8), then @V -> [128,64]
  GA(qh1_bf,768,64,128L*768, kp1,32768,12L*32768, nullptr,
     0,0,0,0, sc1_bf,512,128L*512,12L*128*512, 128,512,64, 0, 12,8);
  softmax_rows<<<dim3(8*12*128), 256, 0, stream>>>(sc1_bf, pr1_bf, 512, 0.125f);
  G(pr1_bf,512,128L*512,12L*128*512, vp1,32768,12L*32768, nullptr,
    0,0,0,0, vattin_bf,768,64,128L*768, 128,64,512, 0, 12,8);
  GA(vattin_bf,768,0,0, ps2o,0,0, s2v_bout, 0,0,0,0, vatt_bf,768,0,0, 1024,768,768, 0, 1,1);

  // ---- question MLP (padded to 16 rows) ----
  G(qpad,768,0,0, ptw1,0,0, t_b1, 0,0,0,0, qmh_bf,768,0,0, 16,768,768, 1, 1,1);
  G(qmh_bf,768,0,0, ptw2,0,0, t_b2, qm_f,768,0,0, 0,0,0,0, 16,768,768, 0, 1,1);

  // ---- answer text / button MLPs (all B*STEPS*A = 1024 rows) ----
  GA(at_bf,768,0,0, ptw1,0,0, t_b1, 0,0,0,0, ath_bf,768,0,0, 1024,768,768, 1, 1,1);
  GA(ath_bf,768,0,0, ptw2,0,0, t_b2, atm_f,768,0,0, 0,0,0,0, 1024,768,768, 0, 1,1);
  GA(ab_bf,768,0,0, pvw1,0,0, v_b1, 0,0,0,0, abh_bf,768,0,0, 1024,768,768, 1, 1,1);
  GA(abh_bf,768,0,0, pvw2,0,0, v_b2, inp_f+2304,3072,0,0, 0,0,0,0, 1024,768,768, 0, 1,1); // ab2 slot

  // qa = q + at  (fp32 into inp slot 3, bf16 for attention A)
  add_qa<<<dim3((1024*768+255)/256), 256, 0, stream>>>(qm_f, atm_f, inp_f, qa_bf);

  // ---- qa->script MHA ----
  GA(qa_bf,768,0,0, pqaq,0,0, qa_bin+0,    0,0,0,0, qh2_bf,768,0,0, 1024,768,768, 0, 1,1);
  GA(sc_bf,768,0,0, pqak,0,0, qa_bin+768,  0,0,0,0, k2_bf,768,0,0,  1024,768,768, 0, 1,1);
  GA(sc_bf,768,0,0, pqav,0,0, qa_bin+1536, 0,0,0,0, v2_bf,768,0,0,  1024,768,768, 0, 1,1);
  PB(k2_bf, 1, 768, 64, 128L*768, kp2, 8192, 12L*8192, 64, 128, 12, 8);
  PB(v2_bf, 768, 1, 64, 128L*768, vp2, 8192, 12L*8192, 128, 64, 12, 8);
  G(qh2_bf,768,64,128L*768, kp2,8192,12L*8192, nullptr,
    0,0,0,0, sc2_bf,128,128L*128,12L*16384, 128,128,64, 0, 12,8);
  softmax_rows<<<dim3(8*12*128), 256, 0, stream>>>(sc2_bf, pr2_bf, 128, 0.125f);
  mean_heads<<<dim3((8*128*128+255)/256), 256, 0, stream>>>(pr2_bf, w_bf);
  G(pr2_bf,128,128L*128,12L*16384, vp2,8192,12L*8192, nullptr,
    0,0,0,0, qsin_bf,768,64,128L*768, 128,64,128, 0, 12,8);
  GA(qsin_bf,768,0,0, pqao,0,0, qa_bout, inp_f+768,3072,0,0, 0,0,0,0, 1024,768,768, 0, 1,1); // qs slot

  // ---- qv = w @ vatt (per batch) -> inp slot 1 ----
  PB(vatt_bf, 768, 1, 0, 128L*768, vattp, 0, 98304, 128, 768, 1, 8);
  GA(w_bf,128,0,128L*128, vattp,0,98304, nullptr,
     inp_f,3072,0,128L*3072, 0,0,0,0, 128,768,128, 0, 1,8);

  // ---- pre MLP: [1024,3072] -> relu -> [1024,768] ----
  CVT(inp_f, inp_bf, 1024L*3072);
  GA(inp_bf,3072,0,0, ppre1,0,0, pre_b1, 0,0,0,0, preh_bf,3072,0,0, 1024,3072,3072, 1, 1,1);
  GA(preh_bf,3072,0,0, ppre2,0,0, pre_b2, inp2_f,768,0,0, inp2_bf,768,0,0, 1024,768,3072, 0, 1,1);

  // gi = inp2 @ wih^T + bih  (state-independent, all rows)
  GA(inp2_bf,768,0,0, pgih,0,0, gru_bih, gi_f,2304,0,0, 0,0,0,0, 1024,2304,768, 0, 1,1);

  // ---- sequential scan: GRU + proj + argmax ----
  init_state<<<dim3(48), 256, 0, stream>>>(state0, state_f, state_bf);
  for (int step = 0; step < 8; ++step) {
    G(state_bf,768,0,0, pghh,0,0, gru_bhh, gh_f,2304,0,0, 0,0,0,0, 16,2304,768, 0, 1,1);
    gru_gates<<<dim3((8*16*768+255)/256), 256, 0, stream>>>(gi_f, gh_f, state_f, h_bf, step);
    GA(h_bf,768,0,0, pprj1,0,0, proj_b1, 0,0,0,0, p1_bf,768,0,0, 128,768,768, 1, 1,1);
    proj_logits<<<dim3(128), 32, 0, stream>>>(p1_bf, proj_w2, proj_b2, out, step);
    argmax_update<<<dim3(8), 32, 0, stream>>>(out, inp2_f, state_f, state_bf, step);
  }
}
